// MultiHeadAttention_71210557768003
// MI455X (gfx1250) — matrix operand order repaired and checked by reading, not compiled, not run on hardware
//
#include <hip/hip_runtime.h>
#include <hip/hip_bf16.h>

#define D_MODEL 1024
#define NHEADS  16
#define HDIM    64
#define SEQ     2048
#define BATCH   2
#define MROWS   (BATCH * SEQ)   // 4096
#define N3      (3 * D_MODEL)   // 3072

typedef __bf16 bf16_t;
typedef __attribute__((ext_vector_type(16))) __bf16   v16bf;
typedef __attribute__((ext_vector_type(8)))  float    v8f;
typedef __attribute__((ext_vector_type(4)))  unsigned u32x4;

union Frag { v16bf v; u32x4 q[2]; };
union Pack8 { unsigned short s[8]; u32x4 v; };

__device__ inline unsigned short f2bf_bits(float f) {
  union { float f; unsigned u; } in; in.f = f;
  unsigned u = in.u;
  u += 0x7fffu + ((u >> 16) & 1u);          // round-to-nearest-even
  return (unsigned short)(u >> 16);
}
__device__ inline bf16_t f2bf(float f) {
  union { unsigned short s; bf16_t h; } o; o.s = f2bf_bits(f); return o.h;
}

__device__ inline v8f wmma_bf16(const Frag& a, const Frag& b, v8f c) {
  // (neg_a, A, neg_b, B, c_mod, C, reuse_a, reuse_b)
  return __builtin_amdgcn_wmma_f32_16x16x32_bf16(false, a.v, false, b.v,
                                                 (short)0, c, false, false);
}

// gfx1250 async copy global -> LDS (per enabled lane: 16 bytes), ASYNCcnt-tracked.
__device__ inline void async_b128_to_lds(unsigned lds_byte_addr, const void* gaddr) {
  asm volatile("global_load_async_to_lds_b128 %0, %1, off"
               :: "v"(lds_byte_addr), "v"(gaddr) : "memory");
}
#define WAIT_ASYNCCNT(n) asm volatile("s_wait_asynccnt " #n ::: "memory")
#define WAIT_DSCNT0()    asm volatile("s_wait_dscnt 0x0" ::: "memory")
// Split workgroup barrier WITHOUT the full loadcnt drain __syncthreads implies:
// cross-wave ordering of the async-LDS stages / ds reads is established by the
// explicit asynccnt/dscnt waits placed before the signal.
#define BAR_SIGNAL()     asm volatile("s_barrier_signal -1" ::: "memory")
#define BAR_WAIT()       asm volatile("s_barrier_wait -1" ::: "memory")
#define MEM_FENCE()      asm volatile("" ::: "memory")

__device__ inline unsigned lds_addr_of(const void* p) {
  return (unsigned)(unsigned long long)p;   // low 32 bits of LDS aperture = offset
}

// ---------------------------------------------------------------- converters
__global__ __launch_bounds__(256) void cvt_f32_bf16(const float* __restrict__ in,
                                                    bf16_t* __restrict__ out, int n) {
  int i = blockIdx.x * blockDim.x + threadIdx.x;
  if (i < n) out[i] = f2bf(in[i]);
}

// in: [K, N] row-major fp32  ->  out: [N, K] row-major bf16
__global__ __launch_bounds__(256) void cvt_transpose(const float* __restrict__ in,
                                                     bf16_t* __restrict__ out,
                                                     int K, int N) {
  int i = blockIdx.x * blockDim.x + threadIdx.x;
  if (i < K * N) {
    int n = i / K, k = i - n * K;
    out[i] = f2bf(in[(size_t)k * N + n]);
  }
}

// ---------------------------------------------------------------- QKV GEMM
// Block: 8 waves, 256(M) x 64(N). B tile (64x32k, 4KB) staged in LDS with
// double-buffered global_load_async_to_lds_b128; each wave does 8 WMMA/k-step.
__global__ __launch_bounds__(256) void qkv_gemm_kernel(
    const bf16_t* __restrict__ xb, const bf16_t* __restrict__ wT,
    const float*  __restrict__ bqkv,
    bf16_t* __restrict__ Qb, bf16_t* __restrict__ Kb, bf16_t* __restrict__ Vt) {
  __shared__ __align__(16) unsigned short bs[2][64 * 32];

  const int tid  = threadIdx.x;
  const int lane = tid & 31;
  const int wave = tid >> 5;
  const int m0   = blockIdx.y * 256 + wave * 32;
  const int n0   = blockIdx.x * 64;
  const int row  = lane & 15;
  const int half = lane >> 4;
  const int ka   = half * 8;    // A-frag K base (ISA 16-bit A layout)
  const int kb   = half * 8;    // B-frag K base (ISA 16-bit B layout)

  // async-stage mapping: 256 threads x 16B = full 4KB B tile per k-step
  const int sn = tid >> 2, sc = tid & 3;
  const bf16_t* gsrc = wT + (size_t)(n0 + sn) * D_MODEL + sc * 8;
  const unsigned l0 = lds_addr_of(&bs[0][sn * 32 + sc * 8]);
  const unsigned l1 = lds_addr_of(&bs[1][sn * 32 + sc * 8]);

  v8f acc[2][4] = {};
  const bf16_t* ap0 = xb + (size_t)(m0 + row) * D_MODEL + ka;
  const bf16_t* ap1 = xb + (size_t)(m0 + 16 + row) * D_MODEL + ka;

  async_b128_to_lds(l0, gsrc);                       // prime buffer 0
  for (int k0 = 0; k0 < D_MODEL; k0 += 32) {
    const int cb = (k0 >> 5) & 1;
    Frag a0, a1;                                     // A from global (per-wave rows)
    a0.q[0] = *(const u32x4*)(ap0 + k0);
    a0.q[1] = *(const u32x4*)(ap0 + k0 + 16);
    a1.q[0] = *(const u32x4*)(ap1 + k0);
    a1.q[1] = *(const u32x4*)(ap1 + k0 + 16);
    __builtin_prefetch(ap0 + k0 + 128, 0, 1);        // global_prefetch_b8
    if (k0 + 32 < D_MODEL) {
      async_b128_to_lds(cb ? l0 : l1, gsrc + k0 + 32);
      WAIT_ASYNCCNT(0x1);                            // current buffer landed
    } else {
      WAIT_ASYNCCNT(0x0);
    }
    BAR_SIGNAL(); BAR_WAIT();                        // all waves' stages visible
                                                     // (A loads stay in flight)
    Frag bf[4];
#pragma unroll
    for (int tt = 0; tt < 4; ++tt) {
      const unsigned short* bp = &bs[cb][(tt * 16 + row) * 32 + kb];
      bf[tt].q[0] = *(const u32x4*)(bp);
      bf[tt].q[1] = *(const u32x4*)(bp + 16);
    }
#pragma unroll
    for (int tt = 0; tt < 4; ++tt) {
      acc[0][tt] = wmma_bf16(a0, bf[tt], acc[0][tt]);
      acc[1][tt] = wmma_bf16(a1, bf[tt], acc[1][tt]);
    }
    WAIT_DSCNT0();                                   // ds reads done ...
    BAR_SIGNAL(); BAR_WAIT();                        // ... before buffer overwrite
  }

  const int h     = n0 / 192;          // head
  const int which = (n0 % 192) / 64;   // 0=Q 1=K 2=V
#pragma unroll
  for (int mm = 0; mm < 2; ++mm) {
#pragma unroll
    for (int t = 0; t < 4; ++t) {
      const int col = t * 16 + row;    // d within head, 0..63
      const float bias = bqkv[n0 + col];
      const int rbase = m0 + mm * 16 + half * 8;   // 8-aligned, batch-aligned
      const int bi = rbase >> 11;
      const int sbase = rbase & (SEQ - 1);
      const int bh = bi * NHEADS + h;
      if (which == 2) {
        Pack8 p;                                    // 8 consecutive s: contiguous in Vt
#pragma unroll
        for (int r = 0; r < 8; ++r) p.s[r] = f2bf_bits(acc[mm][t][r] + bias);
        *(u32x4*)&Vt[((size_t)bh * HDIM + col) * SEQ + sbase] = p.v;
      } else if (which == 0) {
#pragma unroll
        for (int r = 0; r < 8; ++r)
          Qb[((size_t)bh * SEQ + sbase + r) * HDIM + col] =
              f2bf((acc[mm][t][r] + bias) * 0.125f);
      } else {
#pragma unroll
        for (int r = 0; r < 8; ++r)
          Kb[((size_t)bh * SEQ + sbase + r) * HDIM + col] = f2bf(acc[mm][t][r] + bias);
      }
    }
  }
}

// ---------------------------------------------------------------- attention
// One wave per (b*H+h, 16-query tile), flash-style streaming softmax.
// Single-wave workgroup: LDS ops are in-order per wave, so the P staging only
// needs a dscnt fence (no barrier => early V/K loads stay in flight).
__global__ __launch_bounds__(32) void attn_kernel(
    const bf16_t* __restrict__ Qb, const bf16_t* __restrict__ Kb,
    const bf16_t* __restrict__ Vt, bf16_t* __restrict__ ctx) {
  __shared__ __align__(16) unsigned short pls[16 * 32];   // P staging (C->A layout)

  const int lane = threadIdx.x;
  const int bh   = blockIdx.y;
  const int q0   = blockIdx.x * 16;
  const int row  = lane & 15;
  const int half = lane >> 4;
  const int ka   = half * 8;
  const int kb   = half * 8;
  const int kend = q0 + 16;

  const bf16_t* Qh = Qb + (size_t)bh * SEQ * HDIM;
  const bf16_t* Kh = Kb + (size_t)bh * SEQ * HDIM;
  const bf16_t* Vh = Vt + (size_t)bh * HDIM * SEQ;

  Frag qa[2];
#pragma unroll
  for (int c = 0; c < 2; ++c) {
    const bf16_t* qp = Qh + (size_t)(q0 + row) * HDIM + c * 32;
    qa[c].q[0] = *(const u32x4*)(qp + ka);
    qa[c].q[1] = *(const u32x4*)(qp + ka + 16);
  }

  v8f o[4] = {};
  float mrow[8], lrow[8];
#pragma unroll
  for (int r = 0; r < 8; ++r) { mrow[r] = -1e30f; lrow[r] = 0.0f; }

  // kf[j*2+c]: key-subtile j (keys +0/+16), K-chunk c (d 0..31 / 32..63)
  auto loadK = [&](int k0, Frag (&kf)[4]) {
#pragma unroll
    for (int j = 0; j < 2; ++j)
#pragma unroll
      for (int c = 0; c < 2; ++c) {
        const bf16_t* kp = Kh + (size_t)(k0 + j * 16 + row) * HDIM + c * 32 + kb;
        kf[j * 2 + c].q[0] = *(const u32x4*)(kp);
        kf[j * 2 + c].q[1] = *(const u32x4*)(kp + 16);
      }
  };

  auto step = [&](int k0, Frag (&kf)[4], Frag (&nf)[4]) {
    v8f s0 = {}, s1 = {};
    s0 = wmma_bf16(qa[0], kf[0], s0);
    s0 = wmma_bf16(qa[1], kf[1], s0);
    s1 = wmma_bf16(qa[0], kf[2], s1);
    s1 = wmma_bf16(qa[1], kf[3], s1);

    // Issue V loads + next K loads early: they overlap the softmax VALU work.
    Frag vf[4];
#pragma unroll
    for (int t4 = 0; t4 < 4; ++t4) {
      const bf16_t* vp = Vh + (size_t)(t4 * 16 + row) * SEQ + k0 + kb;
      vf[t4].q[0] = *(const u32x4*)(vp);
      vf[t4].q[1] = *(const u32x4*)(vp + 16);
    }
    if (k0 + 32 < kend) loadK(k0 + 32, nf);
    if (k0 + 64 < kend) {
      __builtin_prefetch(Kh + (size_t)(k0 + 64 + row) * HDIM, 0, 1);
      __builtin_prefetch(Vh + (size_t)row * SEQ + k0 + 64, 0, 1);
    }

    const bool need_mask = (k0 + 31) > q0;
#pragma unroll
    for (int r = 0; r < 8; ++r) {
      float a0 = fminf(fmaxf(s0[r], -1e9f), 1e9f);
      float a1 = fminf(fmaxf(s1[r], -1e9f), 1e9f);
      if (need_mask) {
        const int grow = q0 + half * 8 + r;
        if (k0 + row      > grow) a0 = -1e9f;   // clip(-inf) == -1e9 per reference
        if (k0 + 16 + row > grow) a1 = -1e9f;
      }
      float t = fmaxf(a0, a1);
#pragma unroll
      for (int off = 1; off < 16; off <<= 1) t = fmaxf(t, __shfl_xor(t, off, 32));
      const float mnew  = fmaxf(mrow[r], t);
      const float alpha = __expf(mrow[r] - mnew);
      const float p0 = __expf(a0 - mnew);
      const float p1 = __expf(a1 - mnew);
      float rs = p0 + p1;
#pragma unroll
      for (int off = 1; off < 16; off <<= 1) rs += __shfl_xor(rs, off, 32);
      lrow[r] = lrow[r] * alpha + rs;
      mrow[r] = mnew;
#pragma unroll
      for (int t4 = 0; t4 < 4; ++t4) o[t4][r] *= alpha;
      const int prow = half * 8 + r;
      pls[prow * 32 + row]      = f2bf_bits(p0);
      pls[prow * 32 + 16 + row] = f2bf_bits(p1);
    }
    WAIT_DSCNT0();                 // P stores landed (same-wave LDS is in-order)

    Frag pa;   // A-fragment of P (16x32) from LDS
    pa.q[0] = *(const u32x4*)&pls[row * 32 + ka];
    pa.q[1] = *(const u32x4*)&pls[row * 32 + ka + 16];
    MEM_FENCE();                   // keep next iter's stores after these loads

#pragma unroll
    for (int t4 = 0; t4 < 4; ++t4) o[t4] = wmma_bf16(pa, vf[t4], o[t4]);
  };

  Frag kfA[4], kfB[4];
  loadK(0, kfA);
  for (int k0 = 0; k0 < kend; k0 += 64) {
    step(k0, kfA, kfB);
    if (k0 + 32 < kend) step(k0 + 32, kfB, kfA);
  }

  const int b = bh >> 4, h = bh & 15;
#pragma unroll
  for (int r = 0; r < 8; ++r) {
    const float inv = 1.0f / lrow[r];
    const int s = q0 + half * 8 + r;
#pragma unroll
    for (int t4 = 0; t4 < 4; ++t4)
      ctx[((size_t)(b * SEQ + s)) * D_MODEL + h * HDIM + t4 * 16 + row] =
          f2bf(o[t4][r] * inv);
  }
}

// ---------------------------------------------------------------- out GEMM
__global__ __launch_bounds__(256) void out_gemm_kernel(
    const bf16_t* __restrict__ ctx, const bf16_t* __restrict__ woT,
    const float*  __restrict__ bo, float* __restrict__ out) {
  __shared__ __align__(16) unsigned short bs[2][64 * 32];

  const int tid  = threadIdx.x;
  const int lane = tid & 31;
  const int wave = tid >> 5;
  const int m0   = blockIdx.y * 256 + wave * 32;
  const int n0   = blockIdx.x * 64;
  const int row  = lane & 15;
  const int half = lane >> 4;
  const int ka   = half * 8;
  const int kb   = half * 8;

  const int sn = tid >> 2, sc = tid & 3;
  const bf16_t* gsrc = woT + (size_t)(n0 + sn) * D_MODEL + sc * 8;
  const unsigned l0 = lds_addr_of(&bs[0][sn * 32 + sc * 8]);
  const unsigned l1 = lds_addr_of(&bs[1][sn * 32 + sc * 8]);

  v8f acc[2][4] = {};
  const bf16_t* ap0 = ctx + (size_t)(m0 + row) * D_MODEL + ka;
  const bf16_t* ap1 = ctx + (size_t)(m0 + 16 + row) * D_MODEL + ka;

  async_b128_to_lds(l0, gsrc);
  for (int k0 = 0; k0 < D_MODEL; k0 += 32) {
    const int cb = (k0 >> 5) & 1;
    Frag a0, a1;
    a0.q[0] = *(const u32x4*)(ap0 + k0);
    a0.q[1] = *(const u32x4*)(ap0 + k0 + 16);
    a1.q[0] = *(const u32x4*)(ap1 + k0);
    a1.q[1] = *(const u32x4*)(ap1 + k0 + 16);
    __builtin_prefetch(ap0 + k0 + 128, 0, 1);
    if (k0 + 32 < D_MODEL) {
      async_b128_to_lds(cb ? l0 : l1, gsrc + k0 + 32);
      WAIT_ASYNCCNT(0x1);
    } else {
      WAIT_ASYNCCNT(0x0);
    }
    BAR_SIGNAL(); BAR_WAIT();

    Frag bf[4];
#pragma unroll
    for (int tt = 0; tt < 4; ++tt) {
      const unsigned short* bp = &bs[cb][(tt * 16 + row) * 32 + kb];
      bf[tt].q[0] = *(const u32x4*)(bp);
      bf[tt].q[1] = *(const u32x4*)(bp + 16);
    }
#pragma unroll
    for (int tt = 0; tt < 4; ++tt) {
      acc[0][tt] = wmma_bf16(a0, bf[tt], acc[0][tt]);
      acc[1][tt] = wmma_bf16(a1, bf[tt], acc[1][tt]);
    }
    WAIT_DSCNT0();
    BAR_SIGNAL(); BAR_WAIT();
  }
#pragma unroll
  for (int mm = 0; mm < 2; ++mm)
#pragma unroll
    for (int t = 0; t < 4; ++t) {
      const int col = t * 16 + row;
      const float bias = bo[n0 + col];
#pragma unroll
      for (int r = 0; r < 8; ++r) {
        const int grow = m0 + mm * 16 + half * 8 + r;
        out[(size_t)grow * D_MODEL + n0 + col] = acc[mm][t][r] + bias;
      }
    }
}

// ---------------------------------------------------------------- launcher
extern "C" void kernel_launch(void* const* d_in, const int* in_sizes, int n_in,
                              void* d_out, int out_size, void* d_ws, size_t ws_size,
                              hipStream_t stream) {
  const float* x    = (const float*)d_in[0];
  // d_in[1] = mask (bool) — causal mask computed analytically, unused
  const float* Wqkv = (const float*)d_in[2];
  const float* bqkv = (const float*)d_in[3];
  const float* Wo   = (const float*)d_in[4];
  const float* bo   = (const float*)d_in[5];
  float* out = (float*)d_out;

  char* ws = (char*)d_ws;
  const size_t MB = 1u << 20;
  bf16_t* xb    = (bf16_t*)(ws);             // 8 MB  [4096,1024]
  bf16_t* wqkvT = (bf16_t*)(ws +  8 * MB);   // 6 MB  [3072,1024]
  bf16_t* woT   = (bf16_t*)(ws + 14 * MB);   // 2 MB  [1024,1024]
  bf16_t* Qb    = (bf16_t*)(ws + 16 * MB);   // 8 MB  [B,H,S,64] (pre-scaled)
  bf16_t* Kb    = (bf16_t*)(ws + 24 * MB);   // 8 MB  [B,H,S,64]
  bf16_t* Vt    = (bf16_t*)(ws + 32 * MB);   // 8 MB  [B,H,64,S]
  bf16_t* ctx   = xb;                        // reuse: xb dead after qkv_gemm

  const int nx = MROWS * D_MODEL;            // 4,194,304
  cvt_f32_bf16<<<(nx + 255) / 256, 256, 0, stream>>>(x, xb, nx);
  cvt_transpose<<<(D_MODEL * N3 + 255) / 256, 256, 0, stream>>>(Wqkv, wqkvT, D_MODEL, N3);
  cvt_transpose<<<(D_MODEL * D_MODEL + 255) / 256, 256, 0, stream>>>(Wo, woT, D_MODEL, D_MODEL);

  qkv_gemm_kernel<<<dim3(N3 / 64, MROWS / 256), 256, 0, stream>>>(xb, wqkvT, bqkv, Qb, Kb, Vt);
  attn_kernel<<<dim3(SEQ / 16, BATCH * NHEADS), 32, 0, stream>>>(Qb, Kb, Vt, ctx);
  out_gemm_kernel<<<dim3(D_MODEL / 64, MROWS / 256), 256, 0, stream>>>(ctx, woT, bo, out);
}
